// Attention_53944789238513
// MI455X (gfx1250) — compile-verified, matrix operand order checked by reading, not run on hardware
//
#include <hip/hip_runtime.h>
#include <math.h>

#define B_ 32
#define T_ 2048
#define D_ 1024
#define U_ 1024

#define MT 64        // M rows per workgroup in the fused GEMM
#define NCHUNK 64    // N columns processed per chunk
#define KBLK 64      // K block staged in LDS for W1T (double buffered)
#define APAD 8
#define BPAD 8

typedef __bf16 bf16_t;
typedef __attribute__((ext_vector_type(16))) __bf16 v16bf;
typedef __attribute__((ext_vector_type(8)))  __bf16 v8bf;
typedef __attribute__((ext_vector_type(4)))  __bf16 v4bf;
typedef __attribute__((ext_vector_type(8)))  float  v8f;

union V16U { v16bf v; v8bf h[2]; };

// ---- CDNA5 async global->LDS copy (ASYNCcnt-tracked, no VGPR round trip) ----
__device__ __forceinline__ void async_copy_b128(unsigned lds_off, unsigned long long gaddr) {
    asm volatile("global_load_async_to_lds_b128 %0, %1, off"
                 :: "v"(lds_off), "v"(gaddr) : "memory");
}
__device__ __forceinline__ void wait_async0() {
    asm volatile("s_wait_asynccnt 0" ::: "memory");
}

// Branch-free tanh: tanh(x) = 1 - 2/(exp(2x)+1); native exp2 + rcp, saturates
// correctly at +-inf, no exec-mask divergence (ocml tanhf is branchy).
__device__ __forceinline__ float fast_tanh(float x) {
    const float e = __builtin_amdgcn_exp2f(x * 2.88539008177792681472f); // 2*log2(e)
    return 1.0f - 2.0f * __builtin_amdgcn_rcpf(e + 1.0f);
}

// ---------------------------------------------------------------- zero ctx
__global__ void k_zero(float* __restrict__ p, int n) {
    int i = blockIdx.x * 256 + threadIdx.x;
    if (i < n) p[i] = 0.0f;
}

// ------------------------------------------- W1 [D][U] f32 -> W1T [U][D] bf16
__global__ __launch_bounds__(256) void k_w1t(const float* __restrict__ W1,
                                             bf16_t* __restrict__ W1T) {
    __shared__ bf16_t tile[32][33];
    const int tx = threadIdx.x, ty = threadIdx.y;
    const int d0 = blockIdx.y * 32, u0 = blockIdx.x * 32;
#pragma unroll
    for (int i = 0; i < 4; ++i) {
        int d = ty + i * 8;
        tile[d][tx] = (bf16_t)W1[(size_t)(d0 + d) * U_ + u0 + tx];
    }
    __syncthreads();
#pragma unroll
    for (int i = 0; i < 4; ++i) {
        int u = ty + i * 8;
        W1T[(size_t)(u0 + u) * D_ + d0 + tx] = tile[tx][u];
    }
}

// ------------------------------------ projh[b][u] = hidden[b,:]·W2[:,u] + b2[u]
__global__ __launch_bounds__(128) void k_projh(const float* __restrict__ hidden,
                                               const float* __restrict__ W2,
                                               const float* __restrict__ b2,
                                               float* __restrict__ projh) {
    const int u = blockIdx.x * 128 + threadIdx.x;
    float acc[B_];
#pragma unroll
    for (int b = 0; b < B_; ++b) acc[b] = 0.0f;
    for (int d = 0; d < D_; ++d) {
        const float w = W2[(size_t)d * U_ + u];   // coalesced over threads
#pragma unroll
        for (int b = 0; b < B_; ++b) acc[b] += hidden[b * D_ + d] * w;  // s_load broadcast
    }
    const float bb = b2[u];
#pragma unroll
    for (int b = 0; b < B_; ++b) projh[b * U_ + u] = acc[b] + bb;
}

// ---- fused: logits[b,t] = bV + sum_u tanh(features[b,t,:]·W1[:,u] + b1[u] + projh[b,u]) * V[u]
__global__ __launch_bounds__(256) void k_logits(const float* __restrict__ features,
                                                const bf16_t* __restrict__ W1T,
                                                const float* __restrict__ projh,
                                                const float* __restrict__ b1,
                                                const float* __restrict__ Vw,
                                                const float* __restrict__ bV,
                                                float* __restrict__ logits) {
    __shared__ bf16_t A_lds[MT][D_ + APAD];        // 132 KB, full-K A tile (bf16)
    __shared__ bf16_t Bl[2][NCHUNK][KBLK + BPAD];  // 2 x 9 KB, double-buffered W1T block
    __shared__ float  pb_lds[U_];                  // projh + b1
    __shared__ float  vv_lds[U_];                  // V weights
    __shared__ float  lg_lds[MT];                  // per-row logit accumulator

    const int tid  = threadIdx.x;
    const int lane = tid & 31;
    const int wave = tid >> 5;
    const int wm   = wave >> 1;     // 0..3 : which 16-row group
    const int wn   = wave & 1;      // 0..1 : which 32-col half of chunk
    const int nlo  = lane & 15;
    const int h    = lane >> 4;

    const int m0 = blockIdx.x * MT;
    const int b  = m0 / T_;
    const int t0 = m0 % T_;

    // async-copy geometry: 64 rows x 64 bf16 (128 B) per block, 32 B / thread
    const int brow = tid >> 2, bseg = tid & 3;
    const unsigned bl_off0 = (unsigned)(uintptr_t)&Bl[0][brow][bseg * 16];
    const unsigned bl_off1 = (unsigned)(uintptr_t)&Bl[1][brow][bseg * 16];
    const bf16_t* gsrc_base = W1T + (size_t)brow * D_ + bseg * 16;

    const int NKB = D_ / KBLK;                 // 16
    const int NIT = (U_ / NCHUNK) * NKB;       // 256

    auto issue_blk = [&](int buf, int it) {
        const int c0 = (it >> 4) * NCHUNK;     // it / NKB
        const int kb = it & (NKB - 1);
        const unsigned long long g =
            (unsigned long long)(uintptr_t)(gsrc_base + (size_t)c0 * D_ + kb * KBLK);
        const unsigned l = buf ? bl_off1 : bl_off0;
        async_copy_b128(l,      g);
        async_copy_b128(l + 16, g + 16);
    };

    // Prologue: start first W1T block copy, overlapping the A-tile staging below
    issue_blk(0, 0);

    if (tid < MT) lg_lds[tid] = 0.0f;
    for (int u = tid; u < U_; u += 256) {
        pb_lds[u] = projh[b * U_ + u] + b1[u];
        vv_lds[u] = Vw[u];
    }
    // Stage A tile: one feature row (1024 f32) per iteration, convert to bf16
    for (int r = 0; r < MT; ++r) {
        const float4 f = ((const float4*)(features + (size_t)(m0 + r) * D_))[tid];
        v4bf o;
        o[0] = (bf16_t)f.x; o[1] = (bf16_t)f.y; o[2] = (bf16_t)f.z; o[3] = (bf16_t)f.w;
        *(v4bf*)&A_lds[r][tid * 4] = o;
    }

    float rowsum[8];
#pragma unroll
    for (int j = 0; j < 8; ++j) rowsum[j] = 0.0f;
    v8f acc0 = {}; v8f acc1 = {};

    // one pipeline step; buf is a compile-time constant after inlining
    auto step = [&](int it, int buf) {
        const int kb = it & (NKB - 1);
        const int c0 = (it >> 4) * NCHUNK;

        wait_async0();          // current buffer's copy complete (this thread)
        __syncthreads();        // all threads' copies landed; prev-buffer reads done
        if (it + 1 < NIT) issue_blk(buf ^ 1, it + 1);   // prefetch next block

#pragma unroll
        for (int ki = 0; ki < KBLK / 32; ++ki) {
            const int ka  = kb * KBLK + ki * 32;
            const int kbl = ki * 32;
            V16U a, bb0, bb1;
            a.h[0]   = *(const v8bf*)&A_lds[wm * 16 + nlo][ka + h * 8];
            a.h[1]   = *(const v8bf*)&A_lds[wm * 16 + nlo][ka + 16 + h * 8];
            bb0.h[0] = *(const v8bf*)&Bl[buf][wn * 32 + nlo][kbl + h * 8];
            bb0.h[1] = *(const v8bf*)&Bl[buf][wn * 32 + nlo][kbl + 16 + h * 8];
            bb1.h[0] = *(const v8bf*)&Bl[buf][wn * 32 + 16 + nlo][kbl + h * 8];
            bb1.h[1] = *(const v8bf*)&Bl[buf][wn * 32 + 16 + nlo][kbl + 16 + h * 8];
            acc0 = __builtin_amdgcn_wmma_f32_16x16x32_bf16(false, a.v, false, bb0.v,
                                                           (short)0, acc0, false, false);
            acc1 = __builtin_amdgcn_wmma_f32_16x16x32_bf16(false, a.v, false, bb1.v,
                                                           (short)0, acc1, false, false);
        }

        if (kb == NKB - 1) {
            // epilogue: tanh(acc + pb) * V, fold over columns into per-row partials
            const int ca = c0 + wn * 32 + nlo;
            const int cb = ca + 16;
            const float pba = pb_lds[ca], va = vv_lds[ca];
            const float pbb = pb_lds[cb], vb = vv_lds[cb];
#pragma unroll
            for (int j = 0; j < 8; ++j) {
                rowsum[j] += fast_tanh(acc0[j] + pba) * va;
                rowsum[j] += fast_tanh(acc1[j] + pbb) * vb;
            }
            acc0 = (v8f){}; acc1 = (v8f){};
        }
    };

    for (int it = 0; it < NIT; it += 2) {
        step(it,     0);
        step(it + 1, 1);
    }

    // reduce the 16 column-lanes within each half (rows live at lane%16==0)
#pragma unroll
    for (int j = 0; j < 8; ++j) {
        float v = rowsum[j];
        v += __shfl_xor(v, 1, 32);
        v += __shfl_xor(v, 2, 32);
        v += __shfl_xor(v, 4, 32);
        v += __shfl_xor(v, 8, 32);
        rowsum[j] = v;
    }
    if (nlo == 0) {
#pragma unroll
        for (int j = 0; j < 8; ++j)
            atomicAdd(&lg_lds[wm * 16 + h * 8 + j], rowsum[j]);   // combine wn halves
    }
    __syncthreads();
    if (tid < MT) logits[(size_t)b * T_ + t0 + tid] = lg_lds[tid] + bV[0];
}

// --------------------------------------------------- softmax over T per batch
__global__ __launch_bounds__(256) void k_softmax(const float* __restrict__ logits,
                                                 float* __restrict__ attn) {
    __shared__ float red[256];
    const int b = blockIdx.x, tid = threadIdx.x;
    float l[8];
    float mx = -INFINITY;
#pragma unroll
    for (int i = 0; i < 8; ++i) {
        l[i] = logits[b * T_ + tid + i * 256];
        mx = fmaxf(mx, l[i]);
    }
    red[tid] = mx; __syncthreads();
    for (int s = 128; s > 0; s >>= 1) {
        if (tid < s) red[tid] = fmaxf(red[tid], red[tid + s]);
        __syncthreads();
    }
    mx = red[0]; __syncthreads();
    float sum = 0.0f;
#pragma unroll
    for (int i = 0; i < 8; ++i) {
        l[i] = __builtin_amdgcn_exp2f((l[i] - mx) * 1.44269504088896340736f);
        sum += l[i];
    }
    red[tid] = sum; __syncthreads();
    for (int s = 128; s > 0; s >>= 1) {
        if (tid < s) red[tid] += red[tid + s];
        __syncthreads();
    }
    const float inv = 1.0f / red[0];
#pragma unroll
    for (int i = 0; i < 8; ++i) attn[b * T_ + tid + i * 256] = l[i] * inv;
}

// ----------------------- context[b,d] += sum_t attn[b,t] * features[b,t,d]
__global__ __launch_bounds__(256) void k_context(const float* __restrict__ features,
                                                 const float* __restrict__ attn,
                                                 float* __restrict__ ctx) {
    __shared__ float w_lds[64];
    const int b = blockIdx.y;
    const int t0 = blockIdx.x * 64;
    const int tid = threadIdx.x;
    if (tid < 64) w_lds[tid] = attn[b * T_ + t0 + tid];
    __syncthreads();
    float acc[4] = {0.f, 0.f, 0.f, 0.f};
    for (int t = 0; t < 64; ++t) {
        const float w = w_lds[t];
        const float* f = features + (size_t)(b * T_ + t0 + t) * D_;
#pragma unroll
        for (int j = 0; j < 4; ++j) acc[j] += w * f[tid + j * 256];
    }
#pragma unroll
    for (int j = 0; j < 4; ++j)
        atomicAdd(&ctx[b * D_ + tid + j * 256], acc[j]);
}

// ---------------------------------------------------------------------------
extern "C" void kernel_launch(void* const* d_in, const int* in_sizes, int n_in,
                              void* d_out, int out_size, void* d_ws, size_t ws_size,
                              hipStream_t stream) {
    const float* features = (const float*)d_in[0];
    const float* hidden   = (const float*)d_in[1];
    const float* W1       = (const float*)d_in[2];
    const float* b1       = (const float*)d_in[3];
    const float* W2       = (const float*)d_in[4];
    const float* b2       = (const float*)d_in[5];
    const float* Vw       = (const float*)d_in[6];
    const float* bV       = (const float*)d_in[7];

    float* out  = (float*)d_out;
    float* ctx  = out;               // [B, D]  (first output)
    float* attn = out + B_ * D_;     // [B, T, 1] (second output)

    char* ws = (char*)d_ws;
    bf16_t* W1T   = (bf16_t*)ws;                                      // 2 MB
    float*  projh = (float*)(ws + 2u * 1024u * 1024u);                // 128 KB
    float*  logit = (float*)(ws + 2u * 1024u * 1024u + 128u * 1024u); // 256 KB

    k_zero   <<<dim3((B_ * D_ + 255) / 256), dim3(256), 0, stream>>>(ctx, B_ * D_);
    k_w1t    <<<dim3(U_ / 32, D_ / 32), dim3(32, 8), 0, stream>>>(W1, W1T);
    k_projh  <<<dim3(U_ / 128), dim3(128), 0, stream>>>(hidden, W2, b2, projh);
    k_logits <<<dim3(B_ * T_ / MT), dim3(256), 0, stream>>>(features, W1T, projh, b1, Vw, bV, logit);
    k_softmax<<<dim3(B_), dim3(256), 0, stream>>>(logit, attn);
    k_context<<<dim3(T_ / 64, B_), dim3(256), 0, stream>>>(features, attn, ctx);
}
